// AdaptHopLayer_67937792688397
// MI455X (gfx1250) — compile-verified, matrix operand order checked by reading, not run on hardware
//
#include <hip/hip_runtime.h>
#include <hip/hip_bf16.h>

// Problem constants (match reference)
constexpr int N = 2048;
constexpr int Q = 1024;
constexpr int S = 512;
constexpr int NBINS = 20;
constexpr int KT = N / 32;   // 64 k-tiles of 32
constexpr int QT = Q / 16;   // 64 row tiles
constexpr int ST = S / 16;   // 32 col tiles

// Workspace layout (bytes)
constexpr size_t OFF_G1    = 0;                                 // N*N f32 = 16 MB
constexpr size_t OFF_AF    = (size_t)N * N * 4;                 // Q*N bf16 = 4 MB
constexpr size_t OFF_BF    = OFF_AF + (size_t)Q * N * 2;        // N*S bf16 = 2 MB
constexpr size_t OFF_PROBS = OFF_BF + (size_t)N * S * 2;        // 20 f32
constexpr size_t OFF_CWN   = OFF_PROBS + 128;                   // N f32
constexpr size_t OFF_IDL   = OFF_CWN + (size_t)N * 4;           // N f32
constexpr size_t OFF_IDR   = OFF_IDL + (size_t)N * 4;           // N f32
constexpr size_t OFF_STATS = OFF_IDR + (size_t)N * 4;           // 2 f32

typedef __attribute__((ext_vector_type(16))) __bf16 v16bf;
typedef __attribute__((ext_vector_type(8)))  float  v8f;

__device__ __forceinline__ unsigned short f2bf(float f) {
  union { float f; unsigned int u; } v; v.f = f;
  unsigned int u = v.u;
  u += 0x7FFFu + ((u >> 16) & 1u);   // round-to-nearest-even
  return (unsigned short)(u >> 16);
}

// ---- softmax over the 20 calib weights -------------------------------------
__global__ void k_softmax_calib(const float* __restrict__ w, float* __restrict__ probs) {
  if (threadIdx.x == 0) {
    float mx = -3.0e38f;
    for (int i = 0; i < NBINS; ++i) mx = fmaxf(mx, w[i]);
    float e[NBINS]; float s = 0.f;
    for (int i = 0; i < NBINS; ++i) { e[i] = expf(w[i] - mx); s += e[i]; }
    for (int i = 0; i < NBINS; ++i) probs[i] = e[i] / s;
  }
}

// ---- softmax(col_weights) * N ----------------------------------------------
__global__ void k_softmax_cols(const float* __restrict__ w, float* __restrict__ cwN) {
  __shared__ float sm[256];
  __shared__ float mx_s, sum_s;
  int t = threadIdx.x;
  float m = -3.0e38f;
  for (int c = t; c < N; c += 256) m = fmaxf(m, w[c]);
  sm[t] = m; __syncthreads();
  for (int off = 128; off > 0; off >>= 1) {
    if (t < off) sm[t] = fmaxf(sm[t], sm[t + off]);
    __syncthreads();
  }
  if (t == 0) mx_s = sm[0];
  __syncthreads();
  float mx = mx_s;
  float s = 0.f;
  for (int c = t; c < N; c += 256) s += expf(w[c] - mx);
  sm[t] = s; __syncthreads();
  for (int off = 128; off > 0; off >>= 1) {
    if (t < off) sm[t] += sm[t + off];
    __syncthreads();
  }
  if (t == 0) sum_s = sm[0];
  __syncthreads();
  float scale = (float)N / sum_s;
  for (int c = t; c < N; c += 256) cwN[c] = expf(w[c] - mx) * scale;
}

// ---- local Gaussian-bin calibration (all sigmas == 1/950, bases = k/19) ----
__global__ void k_local_calib(const float* __restrict__ g,
                              const float* __restrict__ probs,
                              float* __restrict__ G1) {
  __shared__ float p[NBINS];
  if (threadIdx.x < NBINS) p[threadIdx.x] = probs[threadIdx.x];
  __syncthreads();
  int idx = blockIdx.x * 256 + threadIdx.x;
  float x = g[idx];
  float out = 0.f;
  if (x > 0.f) {
    float zs = 0.f, acc = 0.f;
#pragma unroll
    for (int k = 0; k < NBINS; ++k) {
      float d = x - (float)k * (1.0f / 19.0f);
      float z = expf(-d * d * 950.0f);
      zs += z;
      acc += z * p[k];
    }
    out = (zs > 0.f) ? (acc / zs) : 0.f;
  }
  G1[idx] = out;
}

// ---- degree reductions ------------------------------------------------------
__global__ void k_rowsum(const float* __restrict__ G1, float* __restrict__ inv_dl) {
  __shared__ float sm[256];
  int r = blockIdx.x;
  float s = 0.f;
  for (int c = threadIdx.x; c < N; c += 256) s += G1[(size_t)r * N + c];
  sm[threadIdx.x] = s; __syncthreads();
  for (int off = 128; off > 0; off >>= 1) {
    if (threadIdx.x < off) sm[threadIdx.x] += sm[threadIdx.x + off];
    __syncthreads();
  }
  if (threadIdx.x == 0) {
    float t = sm[0];
    inv_dl[r] = (t > 0.f) ? (1.0f / sqrtf(t)) : 0.f;
  }
}

__global__ void k_colsum(const float* __restrict__ G1, float* __restrict__ inv_dr) {
  int c = blockIdx.x * 256 + threadIdx.x;
  float s = 0.f;
  for (int r = 0; r < N; ++r) s += G1[(size_t)r * N + c];
  inv_dr[c] = (s > 0.f) ? (1.0f / sqrtf(s)) : 0.f;
}

// ---- pack WMMA fragments (gather + scale + bf16) ---------------------------
// 16-bit A 16x32 layout (ISA 7.12.2): lane l (row = l&15), element e:
//   k_local = (e>>3)*16 + ((l>>4)&1)*8 + (e&7)
__global__ void k_packA(const float* __restrict__ G1, const float* __restrict__ inv_dl,
                        const float* __restrict__ inv_dr, const float* __restrict__ cwN,
                        const int* __restrict__ qid, unsigned short* __restrict__ Af) {
  int idx = blockIdx.x * 256 + threadIdx.x;   // Q*N elements
  int e = idx & 15;
  int lane = (idx >> 4) & 31;
  int tile = idx >> 9;
  int kt = tile & (KT - 1);
  int qt = tile >> 6;
  int m = lane & 15;
  int kl = ((e >> 3) << 4) + ((lane >> 4) << 3) + (e & 7);
  int r = qid[qt * 16 + m];
  int k = kt * 32 + kl;
  float v = G1[(size_t)r * N + k] * inv_dl[r] * inv_dr[k] * cwN[k];
  Af[idx] = f2bf(v);
}

__global__ void k_packB(const float* __restrict__ G1, const float* __restrict__ inv_dl,
                        const float* __restrict__ inv_dr, const float* __restrict__ cwN,
                        const int* __restrict__ sid, unsigned short* __restrict__ Bf) {
  int idx = blockIdx.x * 256 + threadIdx.x;   // N*S elements
  int e = idx & 15;
  int lane = (idx >> 4) & 31;
  int tile = idx >> 9;
  int nt = tile & (ST - 1);
  int kt = tile >> 5;
  int n = lane & 15;
  int kl = ((e >> 3) << 4) + ((lane >> 4) << 3) + (e & 7);
  int c = sid[nt * 16 + n];
  int k = kt * 32 + kl;
  float v = G1[(size_t)k * N + c] * inv_dl[k] * inv_dr[c] * cwN[c];
  Bf[idx] = f2bf(v);
}

// ---- WMMA matmul: C[Q,S] = A[Q,N] @ B[N,S], fused diag mask ----------------
__global__ void k_wmma(const unsigned short* __restrict__ Af,
                       const unsigned short* __restrict__ Bf,
                       const int* __restrict__ qid, const int* __restrict__ sid,
                       float* __restrict__ C) {
  const int lane = threadIdx.x & 31;
  const int wave = threadIdx.x >> 5;
  const int tile = blockIdx.x * 8 + wave;
  const int qt = tile >> 5;        // / ST
  const int nt = tile & (ST - 1);

  v8f acc = {};
  const unsigned short* Ap = Af + (((size_t)qt * KT) * 32 + lane) * 16;
  const unsigned short* Bp = Bf + (((size_t)nt) * 32 + lane) * 16;

  for (int kt = 0; kt < KT; ++kt) {
    v16bf a = *(const v16bf*)(const void*)Ap;
    v16bf b = *(const v16bf*)(const void*)Bp;
    acc = __builtin_amdgcn_wmma_f32_16x16x32_bf16(
        /*neg_a=*/false, a, /*neg_b=*/false, b,
        /*c_mod=*/(short)0, acc, /*reuse_a=*/false, /*reuse_b=*/false);
    Ap += 32 * 16;          // next k-tile of this row block
    Bp += (size_t)ST * 32 * 16;
  }

  // C/D layout: VGPR v -> M = v + (lane>=16 ? 8 : 0), N = lane&15
  const int n = lane & 15;
  const int j = nt * 16 + n;
  const int sj = sid[j];
  const int mbase = (lane >> 4) << 3;
#pragma unroll
  for (int v = 0; v < 8; ++v) {
    int i = qt * 16 + mbase + v;
    float val = acc[v];
    if (qid[i] == sj) val = 0.f;   // diagonal mask (i-th query row vs its own node)
    C[(size_t)i * S + j] = val;
  }
}

// ---- global stats over nonzeros of C ---------------------------------------
__global__ void k_stats(const float* __restrict__ C, float* __restrict__ stats) {
  __shared__ float ss[1024];
  __shared__ float sc[1024];
  __shared__ float mean_s;
  int t = threadIdx.x;
  float s = 0.f, cnt = 0.f;
  for (int i = t; i < Q * S; i += 1024) {
    float v = C[i];
    if (v > 0.f) { s += v; cnt += 1.f; }
  }
  ss[t] = s; sc[t] = cnt; __syncthreads();
  for (int off = 512; off > 0; off >>= 1) {
    if (t < off) { ss[t] += ss[t + off]; sc[t] += sc[t + off]; }
    __syncthreads();
  }
  if (t == 0) mean_s = (sc[0] > 0.f) ? (ss[0] / sc[0]) : 0.f;
  __syncthreads();
  float mean = mean_s;
  float vs = 0.f;
  for (int i = t; i < Q * S; i += 1024) {
    float v = C[i];
    if (v > 0.f) { float d = v - mean; vs += d * d; }
  }
  ss[t] = vs; __syncthreads();
  for (int off = 512; off > 0; off >>= 1) {
    if (t < off) ss[t] += ss[t + off];
    __syncthreads();
  }
  if (t == 0) {
    float nnz = sc[0];
    float var = (nnz > 0.f) ? (ss[0] / nnz) : 1.f;
    stats[0] = mean;
    stats[1] = (var > 0.f) ? (1.0f / sqrtf(var)) : 0.f;
  }
}

// ---- global power calibration + elu + row-normalize (in place on d_out) ----
__global__ void k_final(const float* __restrict__ C, const float* __restrict__ w5,
                        const float* __restrict__ b1, const float* __restrict__ stats,
                        float* __restrict__ out) {
  __shared__ float sm[256];
  const int r = blockIdx.x;
  const float mean = stats[0], istd = stats[1];
  const float w0 = w5[0], w1 = w5[1], w2 = w5[2], w3 = w5[3], w4 = w5[4];
  const float b = b1[0];
  float vals[2];
  float s = 0.f;
#pragma unroll
  for (int q = 0; q < 2; ++q) {
    int c = threadIdx.x + q * 256;
    float g = C[(size_t)r * S + c];
    float y = 0.f;
    if (g > 0.f) {
      float gn = (g - mean) * istd;
      float m = fabsf(gn);
      float sq = (m == 0.f) ? 0.f : (gn * sqrtf(m) / m);
      float p2 = gn * m, p3 = p2 * m, p4 = p3 * m;
      float t = w0 * gn + w1 * sq + w2 * p2 + w3 * p3 + w4 * p4 + b;
      float e = (t > 0.f) ? t : (expf(t) - 1.0f);
      y = e + 1.0f;
    }
    vals[q] = y;
    s += y;
  }
  sm[threadIdx.x] = s; __syncthreads();
  for (int off = 128; off > 0; off >>= 1) {
    if (threadIdx.x < off) sm[threadIdx.x] += sm[threadIdx.x + off];
    __syncthreads();
  }
  float rs = sm[0];
  float inv = (rs != 0.f) ? (1.0f / rs) : 0.f;
#pragma unroll
  for (int q = 0; q < 2; ++q) {
    int c = threadIdx.x + q * 256;
    out[(size_t)r * S + c] = vals[q] * inv;
  }
}

extern "C" void kernel_launch(void* const* d_in, const int* in_sizes, int n_in,
                              void* d_out, int out_size, void* d_ws, size_t ws_size,
                              hipStream_t stream) {
  const float* graph    = (const float*)d_in[0];
  const float* calib_w  = (const float*)d_in[1];
  const float* global_w = (const float*)d_in[2];
  const float* global_b = (const float*)d_in[3];
  const float* col_w    = (const float*)d_in[4];
  const int*   qid      = (const int*)d_in[5];
  const int*   sid      = (const int*)d_in[6];

  char* ws = (char*)d_ws;
  float*          G1     = (float*)(ws + OFF_G1);
  unsigned short* Af     = (unsigned short*)(ws + OFF_AF);
  unsigned short* Bf     = (unsigned short*)(ws + OFF_BF);
  float*          probs  = (float*)(ws + OFF_PROBS);
  float*          cwN    = (float*)(ws + OFF_CWN);
  float*          inv_dl = (float*)(ws + OFF_IDL);
  float*          inv_dr = (float*)(ws + OFF_IDR);
  float*          stats  = (float*)(ws + OFF_STATS);
  float*          C      = (float*)d_out;   // Q*S scratch == final output buffer

  k_softmax_calib<<<1, 32, 0, stream>>>(calib_w, probs);
  k_softmax_cols<<<1, 256, 0, stream>>>(col_w, cwN);
  k_local_calib<<<(N * N) / 256, 256, 0, stream>>>(graph, probs, G1);
  k_rowsum<<<N, 256, 0, stream>>>(G1, inv_dl);
  k_colsum<<<N / 256, 256, 0, stream>>>(G1, inv_dr);
  k_packA<<<(Q * N) / 256, 256, 0, stream>>>(G1, inv_dl, inv_dr, cwN, qid, Af);
  k_packB<<<(N * S) / 256, 256, 0, stream>>>(G1, inv_dl, inv_dr, cwN, sid, Bf);
  k_wmma<<<(QT * ST) / 8, 256, 0, stream>>>(Af, Bf, qid, sid, C);
  k_stats<<<1, 1024, 0, stream>>>(C, stats);
  k_final<<<Q, 256, 0, stream>>>(C, global_w, global_b, stats, (float*)d_out);
}
